// CIFFullyDiffAligner_33449205301770
// MI455X (gfx1250) — matrix operand order repaired and checked by reading, not compiled, not run on hardware
//
#include <hip/hip_runtime.h>
#include <hip/hip_bf16.h>

typedef __attribute__((ext_vector_type(16))) _Float16 v16h;
typedef __attribute__((ext_vector_type(8)))  _Float16 v8h;
typedef __attribute__((ext_vector_type(8)))  float    v8f;

#define CB 8
#define CT 4096
#define CD 256
#define CH 128
#define CV 256
#define CNMAX 1024
#define CEPS 1e-8f
#define CLN_EPS 1e-5f

// ---------------------------------------------------------------------------
// Kernel 0: convert f32 weights to f16 in workspace
// ---------------------------------------------------------------------------
__global__ __launch_bounds__(256) void prep_f16_kernel(
    const float* __restrict__ w1, const float* __restrict__ wv,
    _Float16* __restrict__ w1h, _Float16* __restrict__ wvh)
{
    int i = blockIdx.x * blockDim.x + threadIdx.x;
    if (i < CH * CD) w1h[i] = (_Float16)w1[i];
    if (i < CV * CD) wvh[i] = (_Float16)wv[i];
}

// ---------------------------------------------------------------------------
// Kernel 1: per 16-row tile — LN + MLP (WMMA) + alpha raw + v = h*wv^T (WMMA)
// One wave (32 threads) per block; 2048 blocks cover B*T = 32768 rows.
// ---------------------------------------------------------------------------
__global__ __launch_bounds__(32) void fused_mlp_gemm_kernel(
    const float* __restrict__ h,
    const float* __restrict__ gamma, const float* __restrict__ beta,
    const _Float16* __restrict__ w1h, const float* __restrict__ b1,
    const float* __restrict__ w2, const float* __restrict__ b2,
    const _Float16* __restrict__ wvh, const float* __restrict__ bv,
    const int* __restrict__ input_lengths,
    float* __restrict__ v_out,       // [B*T, V] ws
    float* __restrict__ alpha_ws,    // [B*T] masked raw alpha
    float* __restrict__ raw_out)     // [B*T] raw_alpha output
{
    __shared__ float    hf32[16][264];
    __shared__ _Float16 xln [16][264];
    __shared__ _Float16 hh  [16][264];
    __shared__ _Float16 abuf[16][136];

    const int l    = threadIdx.x;
    const int tile = blockIdx.x;
    const int row0 = tile * 16;
    const int b    = row0 >> 12;        // T = 4096
    const int t0   = row0 & (CT - 1);
    const float* hrow = h + (size_t)row0 * CD;

    // prefetch next tile (global_prefetch_b8)
    __builtin_prefetch(hrow + 16 * CD + l * 16, 0, 1);

    // ---- stage tile: f32 copy + f16 copy (for v-GEMM A operand) ----
    for (int j = 0; j < 128; ++j) {
        int idx = j * 32 + l;
        int r = idx >> 8, c0 = idx & 255;
        float x = hrow[idx];
        hf32[r][c0] = x;
        hh[r][c0]   = (_Float16)x;
    }
    __syncthreads();

    const int m  = l & 15;   // row within tile (lanes 0-15 / 16-31 pair up)
    const int hl = l >> 4;   // half selector

    // ---- LayerNorm: lane pair (l, l^16) shares row m ----
    {
        float s1 = 0.f, s2 = 0.f;
        const int cbase = hl * 128;
        for (int i = 0; i < 128; ++i) {
            float x = hf32[m][cbase + i];
            s1 += x; s2 += x * x;
        }
        s1 += __shfl_xor(s1, 16, 32);
        s2 += __shfl_xor(s2, 16, 32);
        float mean = s1 * (1.f / 256.f);
        float var  = s2 * (1.f / 256.f) - mean * mean;
        float rstd = rsqrtf(var + CLN_EPS);
        for (int i = 0; i < 128; ++i) {
            int c = cbase + i;
            float x = hf32[m][c];
            xln[m][c] = (_Float16)(((x - mean) * rstd) * gamma[c] + beta[c]);
        }
    }
    __syncthreads();

    // ---- GEMM1: a = silu(xln[16,256] * w1^T + b1) -> abuf[16,128] f16 ----
    for (int nt = 0; nt < 8; ++nt) {
        v8f acc = {0.f, 0.f, 0.f, 0.f, 0.f, 0.f, 0.f, 0.f};
        for (int kt = 0; kt < 8; ++kt) {
            const int kb = kt * 32 + hl * 8;
            v8h a0 = *(const v8h*)&xln[m][kb];
            v8h a1 = *(const v8h*)&xln[m][kb + 16];
            v16h A = __builtin_shufflevector(a0, a1,
                0,1,2,3,4,5,6,7,8,9,10,11,12,13,14,15);
            const _Float16* bp = w1h + (size_t)(nt * 16 + m) * CD + kt * 32 + hl * 8;
            v16h Bf = __builtin_shufflevector(*(const v8h*)bp, *(const v8h*)(bp + 16),
                0,1,2,3,4,5,6,7,8,9,10,11,12,13,14,15);
            acc = __builtin_amdgcn_wmma_f32_16x16x32_f16(
                false, A, false, Bf, (short)0, acc, false, false);
        }
        const int col = nt * 16 + m;
        const float bias = b1[col];
        for (int r = 0; r < 8; ++r) {
            float x = acc[r] + bias;
            float s = x / (1.f + __expf(-x));   // SiLU
            abuf[r + hl * 8][col] = (_Float16)s;
        }
    }
    __syncthreads();

    // ---- GEMV w2 + softplus -> raw_alpha, masked alpha ----
    {
        float s = 0.f;
        const int hb = hl * 64;
        for (int i = 0; i < 64; ++i)
            s += (float)abuf[m][hb + i] * w2[hb + i];
        s += __shfl_xor(s, 16, 32);
        s += b2[0];
        float raw = (s > 20.f) ? s : log1pf(__expf(s));
        if (l < 16) {
            int t  = t0 + m;
            int gi = b * CT + t;
            raw_out[gi]  = raw;
            alpha_ws[gi] = (t < input_lengths[b]) ? raw : 0.f;
        }
    }

    // ---- GEMM2: v = h[16,256] * wv^T + bv -> v_out ----
    for (int nt = 0; nt < 16; ++nt) {
        v8f acc = {0.f, 0.f, 0.f, 0.f, 0.f, 0.f, 0.f, 0.f};
        for (int kt = 0; kt < 8; ++kt) {
            const int kb = kt * 32 + hl * 8;
            v8h a0 = *(const v8h*)&hh[m][kb];
            v8h a1 = *(const v8h*)&hh[m][kb + 16];
            v16h A = __builtin_shufflevector(a0, a1,
                0,1,2,3,4,5,6,7,8,9,10,11,12,13,14,15);
            const _Float16* bp = wvh + (size_t)(nt * 16 + m) * CD + kt * 32 + hl * 8;
            v16h Bf = __builtin_shufflevector(*(const v8h*)bp, *(const v8h*)(bp + 16),
                0,1,2,3,4,5,6,7,8,9,10,11,12,13,14,15);
            acc = __builtin_amdgcn_wmma_f32_16x16x32_f16(
                false, A, false, Bf, (short)0, acc, false, false);
        }
        const int col  = nt * 16 + m;
        const float bias = bv[col];
        for (int r = 0; r < 8; ++r) {
            int Mr = r + hl * 8;
            v_out[(size_t)(row0 + Mr) * CV + col] = acc[r] + bias;
        }
    }
}

// ---------------------------------------------------------------------------
// Kernel 2: per-batch alpha normalization + inclusive cumsum
// ---------------------------------------------------------------------------
__global__ __launch_bounds__(1024) void scan_alpha_kernel(
    const int* __restrict__ tgt, const int* __restrict__ inl,
    float* __restrict__ alpha_ws, float* __restrict__ c_ws,
    float* __restrict__ alpha_out)
{
    __shared__ float s[1024];
    const int b = blockIdx.x, tid = threadIdx.x;
    const int base = b * CT + tid * 4;

    float x0 = alpha_ws[base + 0], x1 = alpha_ws[base + 1];
    float x2 = alpha_ws[base + 2], x3 = alpha_ws[base + 3];
    float p0 = x0, p1 = p0 + x1, p2 = p1 + x2, p3 = p2 + x3;

    s[tid] = p3;
    __syncthreads();
    for (int off = 1; off < 1024; off <<= 1) {
        float t_ = 0.f;
        if (tid >= off) t_ = s[tid - off];
        __syncthreads();
        s[tid] += t_;
        __syncthreads();
    }
    const float total = s[1023];
    const float excl  = s[tid] - p3;

    const float N  = (float)tgt[b];
    const int   Li = inl[b];
    const float L  = fmaxf((float)Li, 1.f);
    const float scale  = N / (total + CEPS);
    const bool  useUni = total < 10.f * CEPS;
    const float uni    = N / L;

    float xs[4]  = {x0, x1, x2, x3};
    float pre[4] = {p0, p1, p2, p3};
    for (int j = 0; j < 4; ++j) {
        int t = tid * 4 + j;
        float a, c;
        if (useUni) {
            a = (t < Li) ? uni : 0.f;
            c = uni * fminf((float)(t + 1), (float)Li);
        } else {
            a = xs[j] * scale;
            c = (excl + pre[j]) * scale;
        }
        int gi = b * CT + t;
        alpha_ws[gi]  = a;
        c_ws[gi]      = c;
        alpha_out[gi] = a;
    }
}

// ---------------------------------------------------------------------------
// Kernel 3: CIF token gather — one block per (b, n), one V column per thread
// ---------------------------------------------------------------------------
__global__ __launch_bounds__(256) void tokens_kernel(
    const float* __restrict__ c_ws, const float* __restrict__ alpha_ws,
    const float* __restrict__ v_ws, const int* __restrict__ tgt,
    float* __restrict__ tok, float* __restrict__ fd,
    float* __restrict__ mass, float* __restrict__ tmask)
{
    const int blk = blockIdx.x;
    const int b   = blk >> 10;           // NMAX = 1024
    const int n   = blk & (CNMAX - 1);
    const int tid = threadIdx.x;

    const float* cb = c_ws     + b * CT;
    const float* ab = alpha_ws + b * CT;
    const float* vb = v_ws + (size_t)b * CT * CV;

    const float nf = (float)n;       // k - 1
    const float kf = nf + 1.f;       // k

    // first frame whose inclusive mass exceeds k-1
    int lo = 0, hi = CT;
    while (lo < hi) {
        int mid = (lo + hi) >> 1;
        if (cb[mid] > nf) hi = mid; else lo = mid + 1;
    }

    float acc = 0.f, ms = 0.f, du = 0.f;
    for (int t = lo; t < CT; ++t) {
        float ct = cb[t];
        float at = ab[t];
        float cp = ct - at;
        if (cp >= kf) break;
        float w = fminf(ct, kf) - fmaxf(cp, nf);
        w = fmaxf(w, 0.f);
        acc += w * vb[(size_t)t * CV + tid];
        ms  += w;
        du  += w / (at + CEPS);
        if (at <= 0.f) break;        // past input_length: no further mass
    }

    const bool tm = n < tgt[b];
    tok[((size_t)(b * CNMAX + n)) * CV + tid] = tm ? acc : 0.f;
    if (tid == 0) {
        int gi = b * CNMAX + n;
        fd[gi]    = tm ? du : 0.f;
        mass[gi]  = tm ? ms : 0.f;
        tmask[gi] = tm ? 1.f : 0.f;
    }
}

// ---------------------------------------------------------------------------
extern "C" void kernel_launch(void* const* d_in, const int* in_sizes, int n_in,
                              void* d_out, int out_size, void* d_ws, size_t ws_size,
                              hipStream_t stream) {
    const float* h     = (const float*)d_in[0];
    const int*   tgt   = (const int*)  d_in[1];
    const int*   inl   = (const int*)  d_in[2];
    const float* gamma = (const float*)d_in[3];
    const float* beta  = (const float*)d_in[4];
    const float* w1    = (const float*)d_in[5];
    const float* b1    = (const float*)d_in[6];
    const float* w2    = (const float*)d_in[7];
    const float* b2    = (const float*)d_in[8];
    const float* wv    = (const float*)d_in[9];
    const float* bv    = (const float*)d_in[10];

    // workspace layout (bytes)
    char* ws = (char*)d_ws;
    float*    v_ws     = (float*)(ws);                                  // 33,554,432 B
    float*    alpha_ws = (float*)(ws + 33554432);                       //    131,072 B
    float*    c_ws     = (float*)(ws + 33554432 + 131072);              //    131,072 B
    _Float16* w1h      = (_Float16*)(ws + 33554432 + 2 * 131072);       //     65,536 B
    _Float16* wvh      = (_Float16*)(ws + 33554432 + 2 * 131072 + 65536); // 131,072 B

    // output layout (floats, reference tuple order)
    float* out       = (float*)d_out;
    float* tok_out   = out;                              // B*NMAX*V
    float* fd_out    = out + 2097152;                    // B*NMAX
    float* mass_out  = out + 2097152 + 8192;             // B*NMAX
    float* tmask_out = out + 2097152 + 2 * 8192;         // B*NMAX
    float* alpha_out = out + 2097152 + 3 * 8192;         // B*T
    float* raw_out   = out + 2097152 + 3 * 8192 + 32768; // B*T

    // 0) f16 weight conversion
    prep_f16_kernel<<<(CV * CD + 255) / 256, 256, 0, stream>>>(w1, wv, w1h, wvh);

    // 1) fused LN + MLP + value GEMM (WMMA), one wave per 16-row tile
    fused_mlp_gemm_kernel<<<(CB * CT) / 16, 32, 0, stream>>>(
        h, gamma, beta, w1h, b1, w2, b2, wvh, bv, inl,
        v_ws, alpha_ws, raw_out);

    // 2) per-batch normalization + cumsum
    scan_alpha_kernel<<<CB, 1024, 0, stream>>>(tgt, inl, alpha_ws, c_ws, alpha_out);

    // 3) CIF gather per (b, n)
    tokens_kernel<<<CB * CNMAX, 256, 0, stream>>>(
        c_ws, alpha_ws, v_ws, tgt, tok_out, fd_out, mass_out, tmask_out);
}
